// GraphDistRecompAttentionLayer_74045236183107
// MI455X (gfx1250) — matrix operand order repaired and checked by reading, not compiled, not run on hardware
//
#include <hip/hip_runtime.h>
#include <hip/hip_bf16.h>
#include <stddef.h>

// ---- problem constants (match reference) ----
#define N_NODES 8192
#define D_IN    512
#define D_OUT   64
#define ALPHA   0.2f
#define NEG_BIG (-9.0e15f)

typedef __attribute__((ext_vector_type(16))) __bf16 v16bf;
typedef __attribute__((ext_vector_type(8)))  float  v8f;

__device__ __forceinline__ float lrelu(float x) {
    return x > 0.0f ? x : ALPHA * x;
}

// ---------------------------------------------------------------------------
// Kernel 1: Wh = h @ W  via v_wmma_f32_16x16x32_bf16.
// Grid: 512 blocks x 128 threads (4 waves). Block owns rows [i0, i0+16);
// wave w owns output cols [16w, 16w+16). K loop over D_IN in steps of 32.
// Writes Wh (f32, for the rank-1 projections) and Whb (bf16, GEMM-2 operand).
// ---------------------------------------------------------------------------
__global__ void wh_kernel(const float* __restrict__ h,
                          const float* __restrict__ W,
                          float* __restrict__ Wh,
                          __bf16* __restrict__ Whb) {
    const int i0   = blockIdx.x * 16;
    const int wave = threadIdx.x >> 5;
    const int lane = threadIdx.x & 31;
    const int c0   = wave * 16;

    const int arow = i0 + (lane & 15);        // A-fragment row (ISA 16-bit A layout)
    const int kA   = (lane >> 4) * 16;        // A: K chunk  [kA, kA+16)
    const int kB   = (lane >> 4) * 16;        // B: K chunk  [kB, kB+16)
    const int col  = c0 + (lane & 15);        // B/C/D column

    v8f acc = {};
    for (int k0 = 0; k0 < D_IN; k0 += 32) {
        v16bf va, vb;
        const float* hrow = h + (size_t)arow * D_IN + k0 + kA;
#pragma unroll
        for (int t = 0; t < 8; ++t) {
            va[t]     = (__bf16)hrow[t];
            va[8 + t] = (__bf16)hrow[8 + t];
        }
        const float* wcol = W + (size_t)(k0 + kB) * D_OUT + col;
#pragma unroll
        for (int t = 0; t < 16; ++t) vb[t] = (__bf16)wcol[(size_t)t * D_OUT];

        acc = __builtin_amdgcn_wmma_f32_16x16x32_bf16(
            false, va, false, vb, (short)0, acc, false, false);
    }

#pragma unroll
    for (int r = 0; r < 8; ++r) {
        const int orow = i0 + r + ((lane >> 4) << 3);   // C/D layout: +8 for lanes 16..31
        const float v  = acc[r];
        Wh [(size_t)orow * D_OUT + col] = v;
        Whb[(size_t)orow * D_OUT + col] = (__bf16)v;
    }
}

// ---------------------------------------------------------------------------
// Kernel 2: rank-1 projections  w1 = Wh @ a[:64],  w2 = Wh @ a[64:].
// ---------------------------------------------------------------------------
__global__ void wh12_kernel(const float* __restrict__ Wh,
                            const float* __restrict__ a,
                            float* __restrict__ w1,
                            float* __restrict__ w2) {
    const int row = blockIdx.x * blockDim.x + threadIdx.x;
    if (row >= N_NODES) return;
    const float* r = Wh + (size_t)row * D_OUT;
    float s1 = 0.0f, s2 = 0.0f;
#pragma unroll
    for (int f = 0; f < D_OUT; ++f) {
        const float v = r[f];
        s1 += v * a[f];
        s2 += v * a[D_OUT + f];
    }
    w1[row] = s1;
    w2[row] = s2;
}

// ---------------------------------------------------------------------------
// Kernel 3: global max of w2 (32 KB scan -> 1 float).
// Softmax is shift-invariant, and e_ij = lrelu(w1_i + w2_j) is monotone in
// w2_j, so m_hat_i = lrelu(w1_i + max_j w2_j) upper-bounds every row entry.
// Using m_hat as the softmax shift gives the identical result while removing
// the need to ever scan the mask for row statistics.
// ---------------------------------------------------------------------------
__global__ void w2max_kernel(const float* __restrict__ w2,
                             float* __restrict__ w2max) {
    __shared__ float red[256];
    const int t = threadIdx.x;
    float m = -3.0e38f;
    for (int i = t; i < N_NODES; i += 256) m = fmaxf(m, w2[i]);
    red[t] = m;
    __syncthreads();
    for (int s = 128; s > 0; s >>= 1) {
        if (t < s) red[t] = fmaxf(red[t], red[t + s]);
        __syncthreads();
    }
    if (t == 0) *w2max = red[0];
}

// ---------------------------------------------------------------------------
// Kernel 4 (fused stats + GEMM): out = elu( (P @ Wh) / Z ).
// P = exp(v - m_hat) built on the fly; Z accumulated by the staging threads
// alongside the WMMA pipeline, so the 256 MB mask is streamed from HBM
// EXACTLY ONCE (the problem's bandwidth floor: ~11 us at 23.3 TB/s).
// Grid: 512 blocks x 128 threads (4 waves). Block owns rows [i0, i0+16) and
// all 64 output cols. Double-buffered 16x32 LDS weight tile -> one barrier
// per K-step (staging buf b^1 overlaps consumption of buf b; the barrier's
// implicit LDS-count drain protects the buffer before its next rewrite).
// ---------------------------------------------------------------------------
__global__ void attn_out_kernel(const float* __restrict__ mask,
                                const __bf16* __restrict__ Whb,
                                const float* __restrict__ w1,
                                const float* __restrict__ w2,
                                const float* __restrict__ w2maxp,
                                float* __restrict__ out) {
    const int i0   = blockIdx.x * 16;
    const int wave = threadIdx.x >> 5;
    const int lane = threadIdx.x & 31;
    const int c0   = wave * 16;

    __shared__ float pt[2][16][32];   // double-buffered unnormalized weights
    __shared__ float w1s[16], ms[16];
    __shared__ float zpart[16][8];    // per-staging-thread Z partials
    __shared__ float zs[16];          // reduced row sums Z_i

    const float w2max = *w2maxp;
    if (threadIdx.x < 16) {
        const float w1i = w1[i0 + threadIdx.x];
        w1s[threadIdx.x] = w1i;
        ms [threadIdx.x] = lrelu(w1i + w2max);   // m_hat >= all row entries
    }
    __syncthreads();

    // cooperative staging coords: 128 threads x float4 = 16x32 tile
    const int sr = threadIdx.x >> 3;          // staging row 0..15
    const int sc = (threadIdx.x & 7) * 4;     // staging col 0,4,..,28

    // WMMA fragment coords (contiguous K order: lane half h holds k = 16h + s)
    const int ar  = lane & 15;                // A row within tile
    const int kA  = (lane >> 4) * 16;         // A K chunk [kA,kA+16)
    const int kB  = (lane >> 4) * 16;         // B K chunk
    const int col = c0 + (lane & 15);         // output column

    const float* mrowp = mask + (size_t)(i0 + sr) * N_NODES;
    const float  w1i   = w1s[sr];
    const float  mi    = ms[sr];
    float zloc = 0.0f;

    v8f acc = {};
#pragma unroll 2
    for (int j0 = 0; j0 < N_NODES; j0 += 32) {
        const int buf = (j0 >> 5) & 1;

        // prefetch the mask stream ahead (global_prefetch_b8)
        __builtin_prefetch(mrowp + j0 + 256, 0, 1);

        // ---- stage p tile: p = exp(v - m_hat) in [0,1]; masked -> 0 ----
        const float4 mk = *(const float4*)(mrowp + j0 + sc);
        const float4 wv = *(const float4*)(w2 + j0 + sc);
        float v, p;
        v = mk.x > 0.0f ? lrelu(w1i + wv.x) : NEG_BIG;
        p = __expf(v - mi); zloc += p; pt[buf][sr][sc + 0] = p;
        v = mk.y > 0.0f ? lrelu(w1i + wv.y) : NEG_BIG;
        p = __expf(v - mi); zloc += p; pt[buf][sr][sc + 1] = p;
        v = mk.z > 0.0f ? lrelu(w1i + wv.z) : NEG_BIG;
        p = __expf(v - mi); zloc += p; pt[buf][sr][sc + 2] = p;
        v = mk.w > 0.0f ? lrelu(w1i + wv.w) : NEG_BIG;
        p = __expf(v - mi); zloc += p; pt[buf][sr][sc + 3] = p;
        __syncthreads();   // single barrier per K-step (double buffering)

        // ---- A fragment (bf16) from LDS, B fragment from L2-resident Whb ----
        v16bf va, vb;
#pragma unroll
        for (int t = 0; t < 8; ++t) {
            va[t]     = (__bf16)pt[buf][ar][kA + t];
            va[8 + t] = (__bf16)pt[buf][ar][kA + 8 + t];
        }
        const __bf16* bcol = Whb + (size_t)(j0 + kB) * D_OUT + col;
#pragma unroll
        for (int t = 0; t < 16; ++t) vb[t] = bcol[(size_t)t * D_OUT];

        acc = __builtin_amdgcn_wmma_f32_16x16x32_bf16(
            false, va, false, vb, (short)0, acc, false, false);
    }

    // ---- reduce Z per row from the 8 staging threads of that row ----
    zpart[sr][threadIdx.x & 7] = zloc;
    __syncthreads();
    if (threadIdx.x < 16) {
        float z = 0.0f;
#pragma unroll
        for (int q = 0; q < 8; ++q) z += zpart[threadIdx.x][q];
        zs[threadIdx.x] = fmaxf(z, 1.0e-30f);   // guard degenerate empty row
    }
    __syncthreads();

    // ---- epilogue: normalize by Z, ELU, store ----
#pragma unroll
    for (int r = 0; r < 8; ++r) {
        const int rr = r + ((lane >> 4) << 3);
        float vv = acc[r] / zs[rr];
        vv = vv > 0.0f ? vv : (__expf(vv) - 1.0f);
        out[(size_t)(i0 + rr) * D_OUT + col] = vv;
    }
}

// ---------------------------------------------------------------------------
// Launcher. Inputs (setup_inputs order): h, mask, lamda (dead), W, a.
// Workspace: Wh f32 (2 MiB) | Whb bf16 (1 MiB) | w1,w2 (32 KiB each) | w2max.
// ---------------------------------------------------------------------------
extern "C" void kernel_launch(void* const* d_in, const int* in_sizes, int n_in,
                              void* d_out, int out_size, void* d_ws, size_t ws_size,
                              hipStream_t stream) {
    const float* h    = (const float*)d_in[0];
    const float* mask = (const float*)d_in[1];
    // d_in[2] = lamda: dead in the reference dataflow (dischange == 0)
    const float* W    = (const float*)d_in[3];
    const float* a    = (const float*)d_in[4];
    float* out        = (float*)d_out;

    char* ws = (char*)d_ws;
    float*  Wh    = (float*)ws;                                   // 8192*64 f32
    __bf16* Whb   = (__bf16*)(ws + (size_t)N_NODES * D_OUT * 4);  // 8192*64 bf16
    float*  w1    = (float*)(ws + (size_t)N_NODES * D_OUT * 6);
    float*  w2    = w1 + N_NODES;
    float*  w2max = w2 + N_NODES;

    wh_kernel      <<<N_NODES / 16, 128, 0, stream>>>(h, W, Wh, Whb);
    wh12_kernel    <<<N_NODES / 256, 256, 0, stream>>>(Wh, a, w1, w2);
    w2max_kernel   <<<1, 256, 0, stream>>>(w2, w2max);
    attn_out_kernel<<<N_NODES / 16, 128, 0, stream>>>(mask, Whb, w1, w2, w2max, out);
}
